// GeoGNNModel_38104949850639
// MI455X (gfx1250) — matrix operand order repaired and checked by reading, not compiled, not run on hardware
//
#include <hip/hip_runtime.h>
#include <hip/hip_bf16.h>
#include <cmath>

#define NATOMS 30000
#define NBONDS 100000
#define NHALF  50000
#define NANG   200000
#define DD 128
#define HH 256
#define NL 8
#define KB_ 20
#define KA_ 32

typedef __attribute__((ext_vector_type(16))) __bf16 v16bf;
typedef __attribute__((ext_vector_type(8)))  float  v8f;
typedef __attribute__((ext_vector_type(4)))  unsigned int v4u;
struct V4x2 { v4u a, b; };

// ---------------- RBF tables (exp once, reused all layers) ----------------
__global__ void rbf_bond_kernel(const float* __restrict__ x, float* __restrict__ r, int n) {
    int t = blockIdx.x * blockDim.x + threadIdx.x;
    int e = t / KB_;
    if (e >= n) return;
    int k = t - e * KB_;
    float d = x[e] - 0.1f * (float)k;
    r[t] = __expf(-10.0f * d * d);
}

__global__ void rbf_angle_kernel(const float* __restrict__ x, float* __restrict__ r, int n) {
    int t = blockIdx.x * blockDim.x + threadIdx.x;
    int e = t >> 5;
    if (e >= n) return;
    int k = t & 31;
    float d = x[e] - 0.1f * (float)k;
    r[t] = __expf(-10.0f * d * d);
}

// ---------------- embeddings ----------------
__global__ void node_init_kernel(const int* __restrict__ cat, const float* __restrict__ emb,
                                 float* __restrict__ node) {
    int t = blockIdx.x * blockDim.x + threadIdx.x;
    int i = t >> 5;
    if (i >= NATOMS) return;
    int d4 = (t & 31) << 2;
    float ax = 0.f, ay = 0.f, az = 0.f, aw = 0.f;
#pragma unroll
    for (int c = 0; c < 9; ++c) {
        int cc = cat[i * 9 + c];
        const float4 v = *(const float4*)(emb + (((size_t)(c * 64 + cc)) << 7) + d4);
        ax += v.x; ay += v.y; az += v.z; aw += v.w;
    }
    float4 o; o.x = ax; o.y = ay; o.z = az; o.w = aw;
    *(float4*)(node + ((size_t)i << 7) + d4) = o;
}

// bond feat-embed + rbf projection; estride=1 -> all bonds, estride=2 -> even bonds
__global__ void bond_embed_kernel(const int* __restrict__ bcat, const float* __restrict__ rb,
                                  const float* __restrict__ emb, const float* __restrict__ w,
                                  const float* __restrict__ bias, float* __restrict__ out,
                                  int n, int estride) {
    int t = blockIdx.x * blockDim.x + threadIdx.x;
    int i = t >> 5;
    if (i >= n) return;
    int d4 = (t & 31) << 2;
    int e = i * estride;
    float4 bv = *(const float4*)(bias + d4);
    float ax = bv.x, ay = bv.y, az = bv.z, aw = bv.w;
#pragma unroll
    for (int c = 0; c < 3; ++c) {
        int cc = bcat[e * 3 + c];
        const float4 v = *(const float4*)(emb + (((size_t)(c * 16 + cc)) << 7) + d4);
        ax += v.x; ay += v.y; az += v.z; aw += v.w;
    }
    const float* re = rb + (size_t)e * KB_;
#pragma unroll
    for (int k = 0; k < KB_; ++k) {
        float rk = re[k];
        const float4 v = *(const float4*)(w + (k << 7) + d4);
        ax = fmaf(rk, v.x, ax); ay = fmaf(rk, v.y, ay);
        az = fmaf(rk, v.z, az); aw = fmaf(rk, v.w, aw);
    }
    float4 o; o.x = ax; o.y = ay; o.z = az; o.w = aw;
    *(float4*)(out + ((size_t)i << 7) + d4) = o;
}

// ---------------- message scatter (segment-sum via f32 atomics) ----------------
// emsg row index = e >> eshift  (eshift=1 lets us read half-bond rows directly,
// avoiding materializing edge = repeat(half, 2) every layer)
__global__ void scatter_ab_kernel(const float* __restrict__ node, const float* __restrict__ emsg,
                                  int eshift, const int* __restrict__ src, const int* __restrict__ dst,
                                  float* __restrict__ agg, int nE) {
    int t = blockIdx.x * blockDim.x + threadIdx.x;
    int e = t >> 5;
    if (e >= nE) return;
    int d4 = (t & 31) << 2;
    int s = src[e], dd = dst[e];
    const float4 a = *(const float4*)(node + ((size_t)s << 7) + d4);
    const float4 b = *(const float4*)(emsg + ((size_t)(e >> eshift) << 7) + d4);
    float* o = agg + ((size_t)dd << 7) + d4;
    atomicAdd(o + 0, a.x + b.x);
    atomicAdd(o + 1, a.y + b.y);
    atomicAdd(o + 2, a.z + b.z);
    atomicAdd(o + 3, a.w + b.w);
}

// angle-RBF projection fused into scatter
__global__ void scatter_ba_kernel(const float* __restrict__ bond_half, const float* __restrict__ ra,
                                  const float* __restrict__ w, const float* __restrict__ bias,
                                  const int* __restrict__ src, const int* __restrict__ dst,
                                  float* __restrict__ agg, int nE) {
    int t = blockIdx.x * blockDim.x + threadIdx.x;
    int e = t >> 5;
    if (e >= nE) return;
    int d4 = (t & 31) << 2;
    float4 bv = *(const float4*)(bias + d4);
    float ax = bv.x, ay = bv.y, az = bv.z, aw = bv.w;
    const float* re = ra + ((size_t)e << 5);
#pragma unroll
    for (int k = 0; k < KA_; ++k) {
        float rk = re[k];
        const float4 v = *(const float4*)(w + (k << 7) + d4);
        ax = fmaf(rk, v.x, ax); ay = fmaf(rk, v.y, ay);
        az = fmaf(rk, v.z, az); aw = fmaf(rk, v.w, aw);
    }
    int s = src[e], dd = dst[e];
    const float4 h = *(const float4*)(bond_half + ((size_t)s << 7) + d4);
    float* o = agg + ((size_t)dd << 7) + d4;
    atomicAdd(o + 0, ax + h.x);
    atomicAdd(o + 1, ay + h.y);
    atomicAdd(o + 2, az + h.z);
    atomicAdd(o + 3, aw + h.w);
}

// ---------------- conversions ----------------
__global__ void cvt_f32_bf16_kernel(const float* __restrict__ in, __hip_bfloat16* __restrict__ out, int n4) {
    int t = blockIdx.x * blockDim.x + threadIdx.x;
    if (t >= n4) return;
    const float4 v = ((const float4*)in)[t];
    out[4 * (size_t)t + 0] = __float2bfloat16(v.x);
    out[4 * (size_t)t + 1] = __float2bfloat16(v.y);
    out[4 * (size_t)t + 2] = __float2bfloat16(v.z);
    out[4 * (size_t)t + 3] = __float2bfloat16(v.w);
}

// in[R][C] f32 -> out[C][R] bf16 (weights pre-transposed so B frags load contiguously)
__global__ void transpose_bf16_kernel(const float* __restrict__ in, __hip_bfloat16* __restrict__ out,
                                      int R, int C) {
    int t = blockIdx.x * blockDim.x + threadIdx.x;
    if (t >= R * C) return;
    int c = t / R;
    int r = t - c * R;
    out[t] = __float2bfloat16(in[(size_t)r * C + c]);
}

// ---------------- WMMA GEMM: C = A(MxK,bf16) * Bt(NxK,bf16)^T + bias ----------------
__device__ __forceinline__ v16bf load_frag_a(const __hip_bfloat16* A, int row, int K, int kb, int half_) {
    const char* p = (const char*)A + (((size_t)row * K + kb + half_ * 8) << 1);
    v4u lo = *(const v4u*)p;            // K = kb+half*8 .. +7
    v4u hi = *(const v4u*)(p + 32);     // K = kb+16+half*8 .. +7
    return __builtin_bit_cast(v16bf, V4x2{lo, hi});
}

__device__ __forceinline__ v16bf load_frag_b(const __hip_bfloat16* Bt, int col, int K, int kb, int kg) {
    const char* p = (const char*)Bt + (((size_t)col * K + kb + kg * 8) << 1);
    v4u lo = *(const v4u*)p;            // K = kb+kg*8 .. +7
    v4u hi = *(const v4u*)(p + 32);     // K = kb+16+kg*8 .. +7
    return __builtin_bit_cast(v16bf, V4x2{lo, hi});
}

template <bool RELU, bool OUTBF>
__device__ __forceinline__ void gemm_epilogue(const v8f& a0, const v8f& a1, const v8f& a2, const v8f& a3,
                                              int row_base, int tile_n, int l16, int half_,
                                              const float* __restrict__ bias, void* __restrict__ Cout,
                                              int N) {
    v8f accs[4] = {a0, a1, a2, a3};
#pragma unroll
    for (int tt = 0; tt < 4; ++tt) {
        int coln = tile_n + tt * 16 + l16;
        float bv = bias[coln];
#pragma unroll
        for (int r = 0; r < 8; ++r) {
            int rowm = row_base + r + half_ * 8;   // C layout: VGPR r -> M = r (+8 for lanes 16-31)
            float v = accs[tt][r] + bv;
            if (RELU) v = fmaxf(v, 0.0f);
            if (OUTBF)
                ((__hip_bfloat16*)Cout)[(size_t)rowm * N + coln] = __float2bfloat16(v);
            else
                ((float*)Cout)[(size_t)rowm * N + coln] = v;
        }
    }
}

// Each wave: two 16-row bands (row0, row0+64) x 64 cols -> 8 WMMAs share 4 B frags per K-step.
// Row-band guards are forced into SGPRs (readfirstlane) so they lower to scalar branches:
// EXEC is provably all-ones around every WMMA (ISA requirement), no saveexec masking.
template <int K, bool RELU, bool OUTBF>
__global__ __launch_bounds__(128) void gemm_kernel(const __hip_bfloat16* __restrict__ A,
                                                   const __hip_bfloat16* __restrict__ Bt,
                                                   const float* __restrict__ bias,
                                                   void* __restrict__ Cout, int M, int N) {
    int wave  = __builtin_amdgcn_readfirstlane(threadIdx.x >> 5);   // scalar wave id
    int lane  = threadIdx.x & 31;
    int half_ = lane >> 4;
    int l16   = lane & 15;
    int tile_n = blockIdx.x * 64;
    int row0   = blockIdx.y * 128 + wave * 16;   // scalar
    int row1   = row0 + 64;                      // scalar
    if (row0 >= M) return;                       // s_cmp/s_cbranch, EXEC untouched
    const bool do1 = (row1 < M);                 // scalar
    int ar0 = row0 + l16;
    int ar1 = row1 + l16;

    v8f c00 = {}, c01 = {}, c02 = {}, c03 = {};
    v8f c10 = {}, c11 = {}, c12 = {}, c13 = {};
#pragma unroll
    for (int kb = 0; kb < K; kb += 32) {
        v16bf b0 = load_frag_b(Bt, tile_n +  0 + l16, K, kb, half_);
        v16bf b1 = load_frag_b(Bt, tile_n + 16 + l16, K, kb, half_);
        v16bf b2 = load_frag_b(Bt, tile_n + 32 + l16, K, kb, half_);
        v16bf b3 = load_frag_b(Bt, tile_n + 48 + l16, K, kb, half_);
        v16bf af0 = load_frag_a(A, ar0, K, kb, half_);
        c00 = __builtin_amdgcn_wmma_f32_16x16x32_bf16(false, af0, false, b0, (short)0, c00, false, false);
        c01 = __builtin_amdgcn_wmma_f32_16x16x32_bf16(false, af0, false, b1, (short)0, c01, false, false);
        c02 = __builtin_amdgcn_wmma_f32_16x16x32_bf16(false, af0, false, b2, (short)0, c02, false, false);
        c03 = __builtin_amdgcn_wmma_f32_16x16x32_bf16(false, af0, false, b3, (short)0, c03, false, false);
        if (do1) {
            v16bf af1 = load_frag_a(A, ar1, K, kb, half_);
            c10 = __builtin_amdgcn_wmma_f32_16x16x32_bf16(false, af1, false, b0, (short)0, c10, false, false);
            c11 = __builtin_amdgcn_wmma_f32_16x16x32_bf16(false, af1, false, b1, (short)0, c11, false, false);
            c12 = __builtin_amdgcn_wmma_f32_16x16x32_bf16(false, af1, false, b2, (short)0, c12, false, false);
            c13 = __builtin_amdgcn_wmma_f32_16x16x32_bf16(false, af1, false, b3, (short)0, c13, false, false);
        }
    }
    gemm_epilogue<RELU, OUTBF>(c00, c01, c02, c03, row0, tile_n, l16, half_, bias, Cout, N);
    if (do1)
        gemm_epilogue<RELU, OUTBF>(c10, c11, c12, c13, row1, tile_n, l16, half_, bias, Cout, N);
}

// ---------------- fused LayerNorm + scale + relu + residual ----------------
__global__ __launch_bounds__(128) void gnn_post_kernel(const float* __restrict__ h,
                                                       const float* __restrict__ g,
                                                       const float* __restrict__ b,
                                                       const float* __restrict__ resid,
                                                       float* __restrict__ out, float inv_sqrt_m) {
    int row = blockIdx.x;
    int tid = threadIdx.x;
    size_t base = ((size_t)row << 7);
    float x = h[base + tid];
    float s = x, q = x * x;
#pragma unroll
    for (int o = 16; o >= 1; o >>= 1) {
        s += __shfl_xor(s, o, 32);
        q += __shfl_xor(q, o, 32);
    }
    __shared__ float ss[4], qq[4];
    int w = tid >> 5;
    if ((tid & 31) == 0) { ss[w] = s; qq[w] = q; }
    __syncthreads();
    s = ss[0] + ss[1] + ss[2] + ss[3];
    q = qq[0] + qq[1] + qq[2] + qq[3];
    float mu  = s * (1.0f / 128.0f);
    float var = q * (1.0f / 128.0f) - mu * mu;
    float y = (x - mu) * __frsqrt_rn(var + 1e-5f) * g[tid] + b[tid];
    y = fmaxf(y * inv_sqrt_m, 0.0f);
    out[base + tid] = y + resid[base + tid];
}

// edge = repeat(half, 2) -- only materialized once, for the final output
__global__ void repeat2_kernel(const float* __restrict__ half_, float* __restrict__ edge, int n) {
    int t = blockIdx.x * blockDim.x + threadIdx.x;
    int i = t >> 5;
    if (i >= n) return;
    int d4 = (t & 31) << 2;
    const float4 v = *(const float4*)(half_ + ((size_t)i << 7) + d4);
    *(float4*)(edge + ((size_t)(2 * i) << 7) + d4) = v;
    *(float4*)(edge + ((size_t)(2 * i + 1) << 7) + d4) = v;
}

// graph = mean over atoms
__global__ void col_mean_kernel(const float* __restrict__ node, float* __restrict__ graph,
                                int M, float inv) {
    int d  = threadIdx.x & 127;
    int rs = threadIdx.x >> 7;
    int base = blockIdx.x * 256;
    float s = 0.f;
    int lim = base + 256; if (lim > M) lim = M;
    for (int r = base + rs; r < lim; r += 2) s += node[((size_t)r << 7) + d];
    atomicAdd(graph + d, s * inv);
}

// ---------------- host orchestration ----------------
static inline int cdiv(int a, int b) { return (a + b - 1) / b; }

extern "C" void kernel_launch(void* const* d_in, const int* in_sizes, int n_in,
                              void* d_out, int out_size, void* d_ws, size_t ws_size,
                              hipStream_t stream) {
    const int*   atom_cat     = (const int*)  d_in[0];
    const int*   bond_cat     = (const int*)  d_in[1];
    const float* bond_float   = (const float*)d_in[2];
    const float* angle_float  = (const float*)d_in[3];
    const int*   ab_src       = (const int*)  d_in[4];
    const int*   ab_dst       = (const int*)  d_in[5];
    const int*   ba_src       = (const int*)  d_in[6];
    const int*   ba_dst       = (const int*)  d_in[7];
    const float* atom_emb     = (const float*)d_in[8];
    const float* bond_emb0    = (const float*)d_in[9];
    const float* bond_rbf_w0  = (const float*)d_in[10];
    const float* bond_rbf_b0  = (const float*)d_in[11];
    const float* l_bond_emb   = (const float*)d_in[12];
    const float* l_bond_rbf_w = (const float*)d_in[13];
    const float* l_bond_rbf_b = (const float*)d_in[14];
    const float* l_ang_rbf_w  = (const float*)d_in[15];
    const float* l_ang_rbf_b  = (const float*)d_in[16];
    const float* ab_w1 = (const float*)d_in[17];
    const float* ab_b1 = (const float*)d_in[18];
    const float* ab_w2 = (const float*)d_in[19];
    const float* ab_b2 = (const float*)d_in[20];
    const float* ab_g  = (const float*)d_in[21];
    const float* ab_b  = (const float*)d_in[22];
    const float* ba_w1 = (const float*)d_in[23];
    const float* ba_b1 = (const float*)d_in[24];
    const float* ba_w2 = (const float*)d_in[25];
    const float* ba_b2 = (const float*)d_in[26];
    const float* ba_g  = (const float*)d_in[27];
    const float* ba_b  = (const float*)d_in[28];

    // outputs live in d_out: node | edge | graph
    float* node  = (float*)d_out;
    float* edge  = node + (size_t)NATOMS * DD;
    float* graph = edge + (size_t)NBONDS * DD;

    // workspace carve (256B aligned sections)
    char* ws = (char*)d_ws;
    size_t off = 0;
    auto carve = [&](size_t bytes) { void* p = ws + off; off += (bytes + 255) & ~(size_t)255; return p; };
    float* r_bond  = (float*)carve((size_t)NBONDS * KB_ * 4);
    float* r_angle = (float*)carve((size_t)NANG  * KA_ * 4);
    float* agg     = (float*)carve((size_t)NHALF * DD * 4);       // reused for atoms (smaller M)
    __hip_bfloat16* aggbf = (__hip_bfloat16*)carve((size_t)NHALF * DD * 2);
    __hip_bfloat16* h1bf  = (__hip_bfloat16*)carve((size_t)NHALF * HH * 2);
    float* h2        = (float*)carve((size_t)NHALF * DD * 4);
    float* bond_half = (float*)carve((size_t)NHALF * DD * 4);
    __hip_bfloat16* wt = (__hip_bfloat16*)carve((size_t)NL * 4 * DD * HH * 2);

    const int T = 256;
    // --- one-time setup ---
    rbf_bond_kernel <<<cdiv(NBONDS * KB_, T), T, 0, stream>>>(bond_float, r_bond, NBONDS);
    rbf_angle_kernel<<<cdiv(NANG  * KA_, T), T, 0, stream>>>(angle_float, r_angle, NANG);
    node_init_kernel<<<cdiv(NATOMS * 32, T), T, 0, stream>>>(atom_cat, atom_emb, node);
    bond_embed_kernel<<<cdiv(NBONDS * 32, T), T, 0, stream>>>(bond_cat, r_bond, bond_emb0,
                                                              bond_rbf_w0, bond_rbf_b0, edge, NBONDS, 1);
    const size_t WMAT = (size_t)DD * HH;     // 32768 elems per matrix
    for (int l = 0; l < NL; ++l) {
        transpose_bf16_kernel<<<128, T, 0, stream>>>(ab_w1 + l * WMAT, wt + (l * 4 + 0) * WMAT, DD, HH);
        transpose_bf16_kernel<<<128, T, 0, stream>>>(ab_w2 + l * WMAT, wt + (l * 4 + 1) * WMAT, HH, DD);
        transpose_bf16_kernel<<<128, T, 0, stream>>>(ba_w1 + l * WMAT, wt + (l * 4 + 2) * WMAT, DD, HH);
        transpose_bf16_kernel<<<128, T, 0, stream>>>(ba_w2 + l * WMAT, wt + (l * 4 + 3) * WMAT, HH, DD);
    }

    const float inv_sqrt_atoms = 1.0f / sqrtf((float)NATOMS);
    const float inv_sqrt_half  = 1.0f / sqrtf((float)NHALF);

    for (int l = 0; l < NL; ++l) {
        // ---- atom-bond GNN ----
        hipMemsetAsync(agg, 0, (size_t)NATOMS * DD * 4, stream);
        if (l == 0)
            scatter_ab_kernel<<<cdiv(NBONDS * 32, T), T, 0, stream>>>(node, edge, 0,
                                                                      ab_src, ab_dst, agg, NBONDS);
        else  // edge == repeat(half,2): read half rows directly (e>>1), no materialization
            scatter_ab_kernel<<<cdiv(NBONDS * 32, T), T, 0, stream>>>(node, h2, 1,
                                                                      ab_src, ab_dst, agg, NBONDS);
        cvt_f32_bf16_kernel<<<cdiv(NATOMS * DD / 4, T), T, 0, stream>>>(agg, aggbf, NATOMS * DD / 4);
        gemm_kernel<DD, true,  true ><<<dim3(HH / 64, cdiv(NATOMS, 128)), 128, 0, stream>>>(
            aggbf, wt + (l * 4 + 0) * WMAT, ab_b1 + l * HH, h1bf, NATOMS, HH);
        gemm_kernel<HH, false, false><<<dim3(DD / 64, cdiv(NATOMS, 128)), 128, 0, stream>>>(
            h1bf, wt + (l * 4 + 1) * WMAT, ab_b2 + l * DD, h2, NATOMS, DD);
        gnn_post_kernel<<<NATOMS, 128, 0, stream>>>(h2, ab_g + l * DD, ab_b + l * DD,
                                                    node, node, inv_sqrt_atoms);

        // ---- bond(half)-angle GNN ----
        bond_embed_kernel<<<cdiv(NHALF * 32, T), T, 0, stream>>>(
            bond_cat, r_bond, l_bond_emb + (size_t)l * 3 * 16 * DD,
            l_bond_rbf_w + (size_t)l * KB_ * DD, l_bond_rbf_b + l * DD, bond_half, NHALF, 2);
        hipMemsetAsync(agg, 0, (size_t)NHALF * DD * 4, stream);
        scatter_ba_kernel<<<cdiv(NANG * 32, T), T, 0, stream>>>(
            bond_half, r_angle, l_ang_rbf_w + (size_t)l * KA_ * DD, l_ang_rbf_b + l * DD,
            ba_src, ba_dst, agg, NANG);
        cvt_f32_bf16_kernel<<<cdiv(NHALF * DD / 4, T), T, 0, stream>>>(agg, aggbf, NHALF * DD / 4);
        gemm_kernel<DD, true,  true ><<<dim3(HH / 64, cdiv(NHALF, 128)), 128, 0, stream>>>(
            aggbf, wt + (l * 4 + 2) * WMAT, ba_b1 + l * HH, h1bf, NHALF, HH);
        gemm_kernel<HH, false, false><<<dim3(DD / 64, cdiv(NHALF, 128)), 128, 0, stream>>>(
            h1bf, wt + (l * 4 + 3) * WMAT, ba_b2 + l * DD, h2, NHALF, DD);
        gnn_post_kernel<<<NHALF, 128, 0, stream>>>(h2, ba_g + l * DD, ba_b + l * DD,
                                                   bond_half, h2, inv_sqrt_half);
        // h2 now holds "half"; consumed directly by next layer's scatter_ab (e>>1)
    }

    // ---- final edge output + graph readout ----
    repeat2_kernel<<<cdiv(NHALF * 32, T), T, 0, stream>>>(h2, edge, NHALF);
    hipMemsetAsync(graph, 0, DD * 4, stream);
    col_mean_kernel<<<cdiv(NATOMS, 256), 256, 0, stream>>>(node, graph, NATOMS, 1.0f / (float)NATOMS);
}
